// GPT_88519275970815
// MI455X (gfx1250) — matrix operand order repaired and checked by reading, not compiled, not run on hardware
//
#include <hip/hip_runtime.h>
#include <hip/hip_bf16.h>
#include <cmath>
#include <cstdint>

typedef __bf16 bf16;
typedef __attribute__((ext_vector_type(16))) __bf16 v16bf;
typedef __attribute__((ext_vector_type(8)))  __bf16 v8bf;
typedef __attribute__((ext_vector_type(8)))  float  v8f;

#define kV 32000
#define kD 768
#define kH 12
#define kL 2
#define kB 2
#define kT 2048
#define kM (kB * kT)   // 4096 token rows

union FragBF {
  v16bf v;
  v8bf  h[2];
};

__device__ __forceinline__ v8f wmma_bf16(const FragBF& a, const FragBF& b, v8f c) {
  // D = A(16x32 bf16) * B(32x16 bf16) + C(16x16 f32)
  return __builtin_amdgcn_wmma_f32_16x16x32_bf16(false, a.v, false, b.v,
                                                 (short)0, c, false, false);
}

// Async copy of one 16-byte chunk global -> LDS (CDNA5 async data path,
// tracked by ASYNCcnt). lds_off is the raw LDS byte offset (low 32 bits of
// the generic shared-memory address per the LDS aperture mapping).
__device__ __forceinline__ void async_b128(uint32_t lds_off, const void* gaddr) {
  asm volatile("global_load_async_to_lds_b128 %0, %1, off"
               :: "v"(lds_off), "v"(gaddr) : "memory");
}
__device__ __forceinline__ void wait_async_le1() {
  asm volatile("s_wait_asynccnt 0x1" ::: "memory");
}

// ---------------------------------------------------------------------------
// Embedding gather + positional add; produces f32 residual stream + bf16 copy
// ---------------------------------------------------------------------------
__global__ __launch_bounds__(256) void embed_kernel(const int* __restrict__ idx,
    const float* __restrict__ emb, const float* __restrict__ pos,
    float* __restrict__ xf, bf16* __restrict__ xb) {
  const int r = blockIdx.x;          // token row 0..kM-1
  const int t = r % kT;
  const int tok = idx[r];
#pragma unroll
  for (int i = 0; i < 3; ++i) {
    const int d = threadIdx.x + 256 * i;
    float v = emb[(size_t)tok * kD + d] + pos[(size_t)t * kD + d];
    xf[(size_t)r * kD + d] = v;
    xb[(size_t)r * kD + d] = (bf16)v;
  }
}

// ---------------------------------------------------------------------------
// Weight convert f32 [K,N] -> bf16 transposed [N,K]
// ---------------------------------------------------------------------------
__global__ __launch_bounds__(256) void convtr_kernel(const float* __restrict__ W,
    bf16* __restrict__ Wt, int K, int N) {
  const long long i = (long long)blockIdx.x * 256 + threadIdx.x;
  if (i >= (long long)K * N) return;
  const int k = (int)(i / N);
  const int n = (int)(i % N);
  Wt[(size_t)n * K + k] = (bf16)W[i];
}

// ---------------------------------------------------------------------------
// WMMA bf16 GEMM: C[M,N] = A[M,K] @ Bt[N,K]^T + bias, f32 accumulate.
// 256 threads = 8 waves; each wave owns a 16x64 strip of a 128x64 block tile.
// B tile (shared by all 8 waves) is double-buffered in LDS via async copies.
// ---------------------------------------------------------------------------
enum { EPI_F32 = 0, EPI_BF16 = 1, EPI_GELU_BF16 = 2, EPI_QKV = 3 };

__device__ __forceinline__ void stage_b_async(const bf16* __restrict__ Bt, int K,
                                              int n0, int kb, bf16* bs, int tid) {
  // 64 rows x 32 bf16 = 4KB = 256 x 16B chunks; 256 threads x 1 chunk
  const int row = tid >> 2;        // 0..63  (column index within tile)
  const int seg = tid & 3;         // 16B segment within the 64B row
  const bf16* g = Bt + (size_t)(n0 + row) * K + kb + seg * 8;
  const uint32_t l = (uint32_t)(uintptr_t)(bs + row * 32 + seg * 8);
  async_b128(l, g);
}

template<int EPI>
__global__ __launch_bounds__(256) void gemm_kernel(const bf16* __restrict__ A,
    const bf16* __restrict__ Bt, const float* __restrict__ bias,
    float* __restrict__ Cf, bf16* __restrict__ Cb, bf16* __restrict__ Vt,
    int M, int N, int K) {
  __shared__ alignas(16) bf16 Bs[2][64 * 32];
  const int tid = threadIdx.x;
  const int lane = tid & 31;
  const int wave = tid >> 5;
  const int hi = lane >> 4;          // half-wave select
  const int ln = lane & 15;
  const int m0 = blockIdx.y * 128 + wave * 16;
  const int n0 = blockIdx.x * 64;
  (void)M;

  const v8f vzero = {0.f, 0.f, 0.f, 0.f, 0.f, 0.f, 0.f, 0.f};
  v8f acc[4] = {vzero, vzero, vzero, vzero};

  const bf16* Arow = A + (size_t)(m0 + ln) * K;

  // Pipelined: stage k-block 0, preload A fragment for k-block 0.
  stage_b_async(Bt, K, n0, 0, &Bs[0][0], tid);
  FragBF a;
  a.h[0] = *(const v8bf*)(Arow + 8 * hi);
  a.h[1] = *(const v8bf*)(Arow + 8 * hi + 16);

  int buf = 0;
  for (int kb = 0; kb < K; kb += 32) {
    const int nk = (kb + 32 < K) ? (kb + 32) : 0;   // dummy re-stage on last iter
    stage_b_async(Bt, K, n0, nk, &Bs[buf ^ 1][0], tid);
    wait_async_le1();          // previous stage (oldest 1 per thread) complete
    __syncthreads();           // ... for every wave in the block

    // Prefetch next A fragment (overlaps with LDS reads + WMMA below)
    FragBF an;
    an.h[0] = *(const v8bf*)(Arow + nk + 8 * hi);
    an.h[1] = *(const v8bf*)(Arow + nk + 8 * hi + 16);

    const bf16* bsbase = &Bs[buf][0] + 8 * hi;
#pragma unroll
    for (int t = 0; t < 4; ++t) {
      const bf16* brow = bsbase + (16 * t + ln) * 32;
      FragBF bfr;
      bfr.h[0] = *(const v8bf*)(brow);      // ds_load_b128
      bfr.h[1] = *(const v8bf*)(brow + 16);  // ds_load_b128
      acc[t] = wmma_bf16(a, bfr, acc[t]);
    }
    __syncthreads();           // protect Bs[buf] from being overwritten
    a = an;
    buf ^= 1;
  }

#pragma unroll
  for (int t = 0; t < 4; ++t) {
    const int col = n0 + 16 * t + ln;
    const float bv = bias[col];
    if (EPI == EPI_QKV && col >= 2 * kD) {
      // V third of the QKV projection: write transposed [B, D, T] so the
      // attention kernel's V B-fragments are contiguous-per-lane. Rows
      // m0+8*hi .. +7 are contiguous tokens -> one packed 16B store.
      const int d = col - 2 * kD;
      const int bb = m0 / kT;
      const int trow = (m0 % kT) + 8 * hi;
      v8bf pk;
#pragma unroll
      for (int j = 0; j < 8; ++j) pk[j] = (bf16)(acc[t][j] + bv);
      *(v8bf*)(Vt + ((size_t)bb * kD + d) * kT + trow) = pk;
    } else {
#pragma unroll
      for (int j = 0; j < 8; ++j) {
        const int row = m0 + j + 8 * hi;    // C layout: row = j + 8*(lane/16)
        float val = acc[t][j] + bv;
        if (EPI == EPI_GELU_BF16)
          val = 0.5f * val * (1.0f + erff(val * 0.70710678118654752f));
        if (EPI == EPI_F32)
          Cf[(size_t)row * N + col] = val;
        else
          Cb[(size_t)row * N + col] = (bf16)val;
      }
    }
  }
}

// ---------------------------------------------------------------------------
// Flash-attention: one wave owns a 16-query tile; streams 32-key blocks.
// qk layout: [B, T, 3D] bf16 (q | k | unused). V: transposed [B, D, T] bf16.
// Output: [B, T, D] bf16. All LDS tiles are wave-private (no barriers).
// All 8 K/V fragments are issued at the top of each iteration so the V loads
// are in flight during the S-WMMAs and the softmax section.
// ---------------------------------------------------------------------------
__global__ __launch_bounds__(128) void attn_kernel(const bf16* __restrict__ qkv,
                                                   const bf16* __restrict__ vt,
                                                   bf16* __restrict__ outp) {
  __shared__ float Sb[4][16][32];
  __shared__ alignas(16) bf16 Pb[4][16][32];
  __shared__ float Al[4][16];
  __shared__ float Li[4][16];

  const int lane = threadIdx.x & 31;
  const int wave = threadIdx.x >> 5;
  const int hi = lane >> 4;
  const int ln = lane & 15;
  const int bh = blockIdx.y;
  const int b = bh / kH;
  const int h = bh % kH;
  const int t0 = (blockIdx.x * 4 + wave) * 16;

  float* S = &Sb[wave][0][0];
  bf16*  P = &Pb[wave][0][0];
  float* Alpha = Al[wave];
  float* Linv  = Li[wave];

  const size_t rstride = 3 * kD;
  const bf16* base = qkv + (size_t)b * kT * rstride;
  const int hoff = h * 64;

  // Q fragments for hd=64 (two 32-wide k-steps), loaded once
  FragBF aq[2];
  {
    const bf16* qrow = base + (size_t)(t0 + ln) * rstride + hoff;
#pragma unroll
    for (int s = 0; s < 2; ++s) {
      aq[s].h[0] = *(const v8bf*)(qrow + 32 * s + 8 * hi);
      aq[s].h[1] = *(const v8bf*)(qrow + 32 * s + 8 * hi + 16);
    }
  }

  const v8f vzero = {0.f, 0.f, 0.f, 0.f, 0.f, 0.f, 0.f, 0.f};
  v8f o[4] = {vzero, vzero, vzero, vzero};
  float mstat = -1e30f;   // running row max  (valid in lanes 0..15, row = lane)
  float lstat = 0.0f;     // running row sum
  const float scale = 0.03608439182435161f;   // 1/sqrt(768) as in reference

  const bf16* vbase = vt + ((size_t)b * kD + hoff) * kT;

  for (int kb = 0; kb <= t0 + 15; kb += 32) {
    // ---- batch-issue all K fragments (4) and V fragments (4) ----
    FragBF bk[2][2];
#pragma unroll
    for (int kt = 0; kt < 2; ++kt) {
      const bf16* krow =
          base + (size_t)(kb + 16 * kt + ln) * rstride + kD + hoff + 8 * hi;
#pragma unroll
      for (int s = 0; s < 2; ++s) {
        bk[kt][s].h[0] = *(const v8bf*)(krow + 32 * s);
        bk[kt][s].h[1] = *(const v8bf*)(krow + 32 * s + 16);
      }
    }
    FragBF bv[4];
#pragma unroll
    for (int t = 0; t < 4; ++t) {
      const bf16* vcol = vbase + (size_t)(16 * t + ln) * kT + kb + 8 * hi;
      bv[t].h[0] = *(const v8bf*)(vcol);
      bv[t].h[1] = *(const v8bf*)(vcol + 16);
    }

    // ---- S = (q @ k^T) * scale, 16x32 tile ----
#pragma unroll
    for (int kt = 0; kt < 2; ++kt) {
      v8f sacc = vzero;
      sacc = wmma_bf16(aq[0], bk[kt][0], sacc);
      sacc = wmma_bf16(aq[1], bk[kt][1], sacc);
#pragma unroll
      for (int j = 0; j < 8; ++j)
        S[(j + 8 * hi) * 32 + 16 * kt + ln] = sacc[j] * scale;
    }

    // ---- online softmax, lane r<16 owns query row t0+r ----
    if (lane < 16) {
      const int q = t0 + lane;
      float sv[32];
      float bmax = -1e30f;
#pragma unroll
      for (int c = 0; c < 32; ++c) {
        float x = S[lane * 32 + c];
        if (kb + c > q) x = -1e30f;          // causal mask
        sv[c] = x;
        bmax = fmaxf(bmax, x);
      }
      const float nm = fmaxf(mstat, bmax);
      const float alpha = __expf(mstat - nm);
      float psum = 0.0f;
#pragma unroll
      for (int c = 0; c < 32; ++c) {
        float pv = (sv[c] <= -1e29f) ? 0.0f : __expf(sv[c] - nm);
        psum += pv;
        P[lane * 32 + c] = (bf16)pv;
      }
      lstat = lstat * alpha + psum;
      mstat = nm;
      Alpha[lane] = alpha;
    }

    // ---- rescale accumulator rows ----
    float ar[8];
#pragma unroll
    for (int j = 0; j < 8; ++j) ar[j] = Alpha[j + 8 * hi];
#pragma unroll
    for (int t = 0; t < 4; ++t)
#pragma unroll
      for (int j = 0; j < 8; ++j) o[t][j] *= ar[j];

    // ---- O += P(16x32) @ V(32x64); V fragments already in registers ----
    FragBF ap;
    ap.h[0] = *(const v8bf*)(P + ln * 32 + 8 * hi);
    ap.h[1] = *(const v8bf*)(P + ln * 32 + 8 * hi + 16);
#pragma unroll
    for (int t = 0; t < 4; ++t) o[t] = wmma_bf16(ap, bv[t], o[t]);
  }

  // ---- normalize and write [B,T,D] bf16 ----
  if (lane < 16) Linv[lane] = 1.0f / lstat;
  float lr[8];
#pragma unroll
  for (int j = 0; j < 8; ++j) lr[j] = Linv[j + 8 * hi];
  bf16* orow = outp + (size_t)(b * kT + t0) * kD + hoff;
#pragma unroll
  for (int t = 0; t < 4; ++t)
#pragma unroll
    for (int j = 0; j < 8; ++j)
      orow[(size_t)(j + 8 * hi) * kD + 16 * t + ln] = (bf16)(o[t][j] * lr[j]);
}

// ---------------------------------------------------------------------------
// Post-norm residual LayerNorm: y = LN(x + r); writes f32 + bf16 streams
// ---------------------------------------------------------------------------
__global__ __launch_bounds__(256) void ln_kernel(const float* __restrict__ X,
    const float* __restrict__ R, const float* __restrict__ g,
    const float* __restrict__ bb, float* __restrict__ Yf, bf16* __restrict__ Yb) {
  __shared__ float rs[8], rq[8], mv[2];
  const int row = blockIdx.x;
  const float* xr = X + (size_t)row * kD;
  const float* rr = R + (size_t)row * kD;
  float v[3];
  float s = 0.f, q = 0.f;
#pragma unroll
  for (int i = 0; i < 3; ++i) {
    const int d = threadIdx.x + 256 * i;
    float t = xr[d] + rr[d];
    v[i] = t; s += t; q += t * t;
  }
#pragma unroll
  for (int o = 16; o > 0; o >>= 1) {
    s += __shfl_xor(s, o, 32);
    q += __shfl_xor(q, o, 32);
  }
  if ((threadIdx.x & 31) == 0) { rs[threadIdx.x >> 5] = s; rq[threadIdx.x >> 5] = q; }
  __syncthreads();
  if (threadIdx.x == 0) {
    float S = 0.f, Q = 0.f;
    for (int i = 0; i < 8; ++i) { S += rs[i]; Q += rq[i]; }
    float mean = S / kD;
    mv[0] = mean;
    mv[1] = rsqrtf(Q / kD - mean * mean + 1e-5f);
  }
  __syncthreads();
  const float mean = mv[0], inv = mv[1];
#pragma unroll
  for (int i = 0; i < 3; ++i) {
    const int d = threadIdx.x + 256 * i;
    float y = (v[i] - mean) * inv * g[d] + bb[d];
    Yf[(size_t)row * kD + d] = y;
    Yb[(size_t)row * kD + d] = (bf16)y;
  }
}

// Final LayerNorm on the last token of each batch only (2 rows)
__global__ __launch_bounds__(256) void ln_last_kernel(const float* __restrict__ X,
    const float* __restrict__ g, const float* __restrict__ bb,
    float* __restrict__ Y) {
  __shared__ float rs[8], rq[8], mv[2];
  const int brow = blockIdx.x;   // batch index
  const float* xr = X + ((size_t)brow * kT + (kT - 1)) * kD;
  float v[3];
  float s = 0.f, q = 0.f;
#pragma unroll
  for (int i = 0; i < 3; ++i) {
    const int d = threadIdx.x + 256 * i;
    float t = xr[d];
    v[i] = t; s += t; q += t * t;
  }
#pragma unroll
  for (int o = 16; o > 0; o >>= 1) {
    s += __shfl_xor(s, o, 32);
    q += __shfl_xor(q, o, 32);
  }
  if ((threadIdx.x & 31) == 0) { rs[threadIdx.x >> 5] = s; rq[threadIdx.x >> 5] = q; }
  __syncthreads();
  if (threadIdx.x == 0) {
    float S = 0.f, Q = 0.f;
    for (int i = 0; i < 8; ++i) { S += rs[i]; Q += rq[i]; }
    float mean = S / kD;
    mv[0] = mean;
    mv[1] = rsqrtf(Q / kD - mean * mean + 1e-5f);
  }
  __syncthreads();
  const float mean = mv[0], inv = mv[1];
#pragma unroll
  for (int i = 0; i < 3; ++i) {
    const int d = threadIdx.x + 256 * i;
    Y[(size_t)brow * kD + d] = (v[i] - mean) * inv * g[d] + bb[d];
  }
}

// ---------------------------------------------------------------------------
// LM head for last token only: logits[b, v] = x_last[b] . Wlm[:, v] + blm[v]
// float4-vectorized along the vocab dimension (4 columns per thread).
// ---------------------------------------------------------------------------
__global__ __launch_bounds__(256) void lmhead_kernel(const float* __restrict__ xl,
    const float* __restrict__ W, const float* __restrict__ bias,
    float* __restrict__ outp) {
  __shared__ float xs[kD];
  const int b = blockIdx.y;
  for (int d = threadIdx.x; d < kD; d += 256) xs[d] = xl[(size_t)b * kD + d];
  __syncthreads();
  const int v4 = (blockIdx.x * 256 + threadIdx.x) * 4;
  if (v4 >= kV) return;
  float4 acc = *(const float4*)(bias + v4);
  for (int d = 0; d < kD; ++d) {
    const float4 w = *(const float4*)(W + (size_t)d * kV + v4);
    const float x = xs[d];
    acc.x += x * w.x; acc.y += x * w.y; acc.z += x * w.z; acc.w += x * w.w;
  }
  *(float4*)(outp + (size_t)b * kV + v4) = acc;
}

// ---------------------------------------------------------------------------
extern "C" void kernel_launch(void* const* d_in, const int* in_sizes, int n_in,
                              void* d_out, int out_size, void* d_ws, size_t ws_size,
                              hipStream_t stream) {
  (void)in_sizes; (void)n_in; (void)out_size; (void)ws_size;
  const int*   idx  = (const int*)d_in[0];
  const float* emb  = (const float*)d_in[1];
  const float* pos  = (const float*)d_in[2];
  const float* Wqkv = (const float*)d_in[3];
  const float* bqkv = (const float*)d_in[4];
  const float* Wo   = (const float*)d_in[5];
  const float* bo   = (const float*)d_in[6];
  const float* g1   = (const float*)d_in[7];
  const float* b1   = (const float*)d_in[8];
  const float* Wfc  = (const float*)d_in[9];
  const float* bfc  = (const float*)d_in[10];
  const float* Wpr  = (const float*)d_in[11];
  const float* bpr  = (const float*)d_in[12];
  const float* g2   = (const float*)d_in[13];
  const float* b2   = (const float*)d_in[14];
  const float* gf   = (const float*)d_in[15];
  const float* bfb  = (const float*)d_in[16];
  const float* Wlm  = (const float*)d_in[17];
  const float* blm  = (const float*)d_in[18];
  float* out = (float*)d_out;

  // Workspace carving (256B aligned)
  char* base = (char*)d_ws;
  size_t off = 0;
  auto carve = [&](size_t bytes) -> char* {
    char* r = base + off;
    off += (bytes + 255) & ~(size_t)255;
    return r;
  };
  float* xf    = (float*)carve((size_t)kM * kD * 4);        // residual stream f32
  bf16*  xb    = (bf16*) carve((size_t)kM * kD * 2);        // bf16 copy for GEMMs
  bf16*  qkvb  = (bf16*) carve((size_t)kM * 3 * kD * 2);    // q|k bf16 ([M,3D], v third unused)
  bf16*  vtb   = (bf16*) carve((size_t)kB * kD * kT * 2);   // V transposed [B, D, T]
  bf16*  attnb = (bf16*) carve((size_t)kM * kD * 2);        // attention out bf16
  float* projf = (float*)carve((size_t)kM * kD * 4);        // proj / ffn out f32
  bf16*  hb    = (bf16*) carve((size_t)kM * 4 * kD * 2);    // gelu hidden bf16
  bf16*  wT    = (bf16*) carve((size_t)4 * kD * kD * 2);    // transposed weight
  float* xlast = (float*)carve((size_t)kB * kD * 4);        // last-token LN out

  embed_kernel<<<kM, 256, 0, stream>>>(idx, emb, pos, xf, xb);

  for (int l = 0; l < kL; ++l) {
    // ---- attention ----
    convtr_kernel<<<(kD * 3 * kD) / 256, 256, 0, stream>>>(
        Wqkv + (size_t)l * kD * 3 * kD, wT, kD, 3 * kD);
    gemm_kernel<EPI_QKV><<<dim3(3 * kD / 64, kM / 128), 256, 0, stream>>>(
        xb, wT, bqkv + (size_t)l * 3 * kD, nullptr, qkvb, vtb, kM, 3 * kD, kD);
    attn_kernel<<<dim3(kT / 64, kB * kH), 128, 0, stream>>>(qkvb, vtb, attnb);
    convtr_kernel<<<(kD * kD) / 256, 256, 0, stream>>>(
        Wo + (size_t)l * kD * kD, wT, kD, kD);
    gemm_kernel<EPI_F32><<<dim3(kD / 64, kM / 128), 256, 0, stream>>>(
        attnb, wT, bo + (size_t)l * kD, projf, nullptr, nullptr, kM, kD, kD);
    ln_kernel<<<kM, 256, 0, stream>>>(xf, projf, g1 + (size_t)l * kD,
                                      b1 + (size_t)l * kD, xf, xb);
    // ---- MLP ----
    convtr_kernel<<<(kD * 4 * kD) / 256, 256, 0, stream>>>(
        Wfc + (size_t)l * kD * 4 * kD, wT, kD, 4 * kD);
    gemm_kernel<EPI_GELU_BF16><<<dim3(4 * kD / 64, kM / 128), 256, 0, stream>>>(
        xb, wT, bfc + (size_t)l * 4 * kD, nullptr, hb, nullptr, kM, 4 * kD, kD);
    convtr_kernel<<<(4 * kD * kD) / 256, 256, 0, stream>>>(
        Wpr + (size_t)l * 4 * kD * kD, wT, 4 * kD, kD);
    gemm_kernel<EPI_F32><<<dim3(kD / 64, kM / 128), 256, 0, stream>>>(
        hb, wT, bpr + (size_t)l * kD, projf, nullptr, nullptr, kM, kD, 4 * kD);
    ln_kernel<<<kM, 256, 0, stream>>>(xf, projf, g2 + (size_t)l * kD,
                                      b2 + (size_t)l * kD, xf, xb);
  }

  ln_last_kernel<<<kB, 256, 0, stream>>>(xf, gf, bfb, xlast);
  lmhead_kernel<<<dim3((kV / 4 + 255) / 256, kB), 256, 0, stream>>>(
      xlast, Wlm, blm, out);
}
